// MultiHeadAttention_8426725835220
// MI455X (gfx1250) — matrix operand order repaired and checked by reading, not compiled, not run on hardware
//
#include <hip/hip_runtime.h>
#include <hip/hip_bf16.h>

typedef __attribute__((ext_vector_type(16))) _Float16 v16h;
typedef __attribute__((ext_vector_type(8)))  _Float16 v8h;
typedef __attribute__((ext_vector_type(8)))  float    v8f;
typedef __attribute__((ext_vector_type(4)))  float    v4f;

#define S_LEN 2048
#define D_DIM 512
#define H_NUM 8
#define HD    64

__device__ __forceinline__ v8f wmma_f16(v16h a, v16h b, v8f c) {
    // D = A(16x32 f16) * B(32x16 f16) + C(16x16 f32)
    return __builtin_amdgcn_wmma_f32_16x16x32_f16(
        /*neg_a=*/false, a, /*neg_b=*/false, b,
        /*c_mod=*/(short)0, c, /*reuse_a=*/false, /*reuse_b=*/false);
}

// ---------------------------------------------------------------------------
// Kernel 0a: convert + transpose the four DxD weight matrices to f16, Wt[n][k]
// ---------------------------------------------------------------------------
__global__ void mha_prep_weights(const float* __restrict__ Wq,
                                 const float* __restrict__ Wk,
                                 const float* __restrict__ Wv,
                                 const float* __restrict__ Wo,
                                 _Float16* __restrict__ Wqt,
                                 _Float16* __restrict__ Wkt,
                                 _Float16* __restrict__ Wvt,
                                 _Float16* __restrict__ Wot) {
    int idx = blockIdx.x * blockDim.x + threadIdx.x;   // 0 .. 4*512*512-1
    int which = idx >> 18;                             // 512*512 = 1<<18
    int e = idx & 262143;
    int k = e >> 9;
    int n = e & 511;
    const float* W = (which == 0) ? Wq : (which == 1) ? Wk : (which == 2) ? Wv : Wo;
    _Float16*    T = (which == 0) ? Wqt : (which == 1) ? Wkt : (which == 2) ? Wvt : Wot;
    T[(size_t)n * D_DIM + k] = (_Float16)W[(size_t)k * D_DIM + n];
}

// ---------------------------------------------------------------------------
// Kernel 0b: convert the three [B*S, D] fp32 activations to f16 (row-major).
// ---------------------------------------------------------------------------
__global__ void mha_prep_x(const float* __restrict__ q,
                           const float* __restrict__ k,
                           const float* __restrict__ v,
                           _Float16* __restrict__ q16,
                           _Float16* __restrict__ k16,
                           _Float16* __restrict__ v16) {
    size_t idx = (size_t)blockIdx.x * blockDim.x + threadIdx.x;  // 8 elems each
    int which = (int)(idx >> 19);            // 4194304/8 = 1<<19 per tensor
    size_t e = (idx & 524287) * 8;
    const float* src = (which == 0) ? q : (which == 1) ? k : v;
    _Float16*    dst = (which == 0) ? q16 : (which == 1) ? k16 : v16;
    v4f a = *(const v4f*)(src + e);
    v4f b = *(const v4f*)(src + e + 4);
    v8h o;
#pragma unroll
    for (int i = 0; i < 4; i++) {
        o[i]     = (_Float16)a[i];
        o[4 + i] = (_Float16)b[i];
    }
    *(v8h*)(dst + e) = o;
}

// ---------------------------------------------------------------------------
// Shared 32(M) x 64(N) per-wave GEMM core, f16 A/B, f32 accumulate.
// 8 accumulators; each B tile reused across 2 M-subtiles (0.75 KB cache
// reads per WMMA vs 1.25 KB at 16x64).  Fully-unrolled k-loop (16 steps).
// acc[mt][nt]: mt = M-subtile (0: rows mBase+0..15, 1: rows mBase+16..31)
// ---------------------------------------------------------------------------
__device__ __forceinline__ void gemm_tile_32x64(const _Float16* __restrict__ Arow0,
                                                const _Float16* __restrict__ Arow1,
                                                const _Float16* __restrict__ Brow0,
                                                const _Float16* __restrict__ Brow1,
                                                const _Float16* __restrict__ Brow2,
                                                const _Float16* __restrict__ Brow3,
                                                int half, v8f acc[2][4]) {
#pragma unroll
    for (int kb = 0; kb < D_DIM; kb += 32) {
        v8h a00 = *(const v8h*)(Arow0 + kb + half * 16);
        v8h a01 = *(const v8h*)(Arow0 + kb + half * 16 + 8);
        v8h a10 = *(const v8h*)(Arow1 + kb + half * 16);
        v8h a11 = *(const v8h*)(Arow1 + kb + half * 16 + 8);
        v16h Av0, Av1;
#pragma unroll
        for (int i = 0; i < 8; i++) {
            Av0[i] = a00[i]; Av0[8 + i] = a01[i];
            Av1[i] = a10[i]; Av1[8 + i] = a11[i];
        }
        v16h B0 = *(const v16h*)(Brow0 + kb);
        v16h B1 = *(const v16h*)(Brow1 + kb);
        v16h B2 = *(const v16h*)(Brow2 + kb);
        v16h B3 = *(const v16h*)(Brow3 + kb);
        acc[0][0] = wmma_f16(Av0, B0, acc[0][0]);
        acc[0][1] = wmma_f16(Av0, B1, acc[0][1]);
        acc[0][2] = wmma_f16(Av0, B2, acc[0][2]);
        acc[0][3] = wmma_f16(Av0, B3, acc[0][3]);
        acc[1][0] = wmma_f16(Av1, B0, acc[1][0]);
        acc[1][1] = wmma_f16(Av1, B1, acc[1][1]);
        acc[1][2] = wmma_f16(Av1, B2, acc[1][2]);
        acc[1][3] = wmma_f16(Av1, B3, acc[1][3]);
    }
}

// ---------------------------------------------------------------------------
// Kernel 1: QKV projection GEMMs. z = 0/1/2 selects Q/K/V (f16 inputs).
// Q,K stored [b,h,s,hd] f16.  V stored transposed [b,h,hd,s] f16.
// ---------------------------------------------------------------------------
__global__ __launch_bounds__(256) void mha_proj_qkv(
    const _Float16* __restrict__ q16, const _Float16* __restrict__ k16,
    const _Float16* __restrict__ v16,
    const _Float16* __restrict__ Wqt, const _Float16* __restrict__ Wkt,
    const _Float16* __restrict__ Wvt,
    const float* __restrict__ bq, const float* __restrict__ bk,
    const float* __restrict__ bv,
    _Float16* __restrict__ Qh, _Float16* __restrict__ Kh,
    _Float16* __restrict__ Vt)
{
    const int z = blockIdx.z;
    const _Float16* X    = (z == 0) ? q16 : (z == 1) ? k16 : v16;
    const _Float16* Wt   = (z == 0) ? Wqt : (z == 1) ? Wkt : Wvt;
    const float*    bias = (z == 0) ? bq  : (z == 1) ? bk  : bv;
    _Float16*       dst  = (z == 0) ? Qh  : (z == 1) ? Kh  : Vt;

    const int lane = threadIdx.x & 31;
    const int wave = threadIdx.x >> 5;
    const int half = lane >> 4;
    const int l16  = lane & 15;
    const int mBase = blockIdx.x * 256 + wave * 32;
    const int nBase = blockIdx.y * 64;

    const _Float16* Arow0 = X + (size_t)(mBase + l16) * D_DIM;
    const _Float16* Arow1 = X + (size_t)(mBase + 16 + l16) * D_DIM;
    const _Float16* Brow0 = Wt + (size_t)(nBase + 0 * 16 + l16) * D_DIM + half * 16;
    const _Float16* Brow1 = Wt + (size_t)(nBase + 1 * 16 + l16) * D_DIM + half * 16;
    const _Float16* Brow2 = Wt + (size_t)(nBase + 2 * 16 + l16) * D_DIM + half * 16;
    const _Float16* Brow3 = Wt + (size_t)(nBase + 3 * 16 + l16) * D_DIM + half * 16;

    v8f acc[2][4] = {};
    gemm_tile_32x64(Arow0, Arow1, Brow0, Brow1, Brow2, Brow3, half, acc);

    // Epilogue: bias add, convert f16, scatter to head-split layout
#pragma unroll
    for (int nt = 0; nt < 4; nt++) {
        int gcol = nBase + nt * 16 + l16;
        float bb = bias[gcol];
        int h  = gcol >> 6;       // / 64
        int dh = gcol & 63;
#pragma unroll
        for (int mt = 0; mt < 2; mt++) {
#pragma unroll
            for (int r = 0; r < 8; r++) {
                int gm = mBase + mt * 16 + r + half * 8;
                int b  = gm >> 11;    // / 2048
                int s  = gm & 2047;
                float val = acc[mt][nt][r] + bb;
                size_t off;
                if (z == 2)
                    off = ((size_t)(b * H_NUM + h) * HD + dh) * S_LEN + s; // Vt[b,h,hd,s]
                else
                    off = ((size_t)(b * H_NUM + h) * S_LEN + s) * HD + dh; // [b,h,s,hd]
                dst[off] = (_Float16)val;
            }
        }
    }
}

// ---------------------------------------------------------------------------
// Kernel 2: flash attention. One wave per (b, h, 16-query tile).
// 32-key blocks: 4 WMMA scores, online softmax (V loads issued above it),
// P staged through LDS into A-operand layout, 4 WMMA for P@V.
// global_prefetch of next block's K/V rows hides next-iteration latency.
// ---------------------------------------------------------------------------
__global__ __launch_bounds__(32) void mha_attention(
    const _Float16* __restrict__ Qh, const _Float16* __restrict__ Kh,
    const _Float16* __restrict__ Vt, _Float16* __restrict__ attn)
{
    __shared__ _Float16 Pl[16][32];

    const int lane = threadIdx.x & 31;
    const int half = lane >> 4;
    const int l16  = lane & 15;
    const int qt = blockIdx.x & 127;
    const int h  = (blockIdx.x >> 7) & 7;
    const int b  = blockIdx.x >> 10;

    const _Float16* Qb = Qh + (size_t)(b * H_NUM + h) * S_LEN * HD;
    const _Float16* Kb = Kh + (size_t)(b * H_NUM + h) * S_LEN * HD;
    const _Float16* Vb = Vt + (size_t)(b * H_NUM + h) * HD * S_LEN;

    // Load Q A-operands for k-chunks [0,32) and [32,64)
    const _Float16* Qr = Qb + (size_t)(qt * 16 + l16) * HD;
    v16h QA[2];
#pragma unroll
    for (int c = 0; c < 2; c++) {
        v8h r0 = *(const v8h*)(Qr + c * 32 + half * 16);
        v8h r1 = *(const v8h*)(Qr + c * 32 + half * 16 + 8);
#pragma unroll
        for (int i = 0; i < 8; i++) { QA[c][i] = r0[i]; QA[c][8 + i] = r1[i]; }
    }

    v8f O[4] = {};
    float mrow[8], lrow[8];
#pragma unroll
    for (int r = 0; r < 8; r++) { mrow[r] = -__builtin_inff(); lrow[r] = 0.f; }

    const int nkb = (qt >> 1) + 1;   // number of 32-key blocks (causal)
    for (int kb2 = 0; kb2 < nkb; kb2++) {
        const int key0 = kb2 * 32;

        // scores: two 16x16 tiles, each chained over two K=32 WMMAs
        v8f Sc[2];
#pragma unroll
        for (int t = 0; t < 2; t++) {
            const _Float16* Krow = Kb + (size_t)(key0 + t * 16 + l16) * HD;
            v16h KB0 = *(const v16h*)(Krow + half * 16);
            v16h KB1 = *(const v16h*)(Krow + 32 + half * 16);
            v8f zero = {};
            v8f s0 = wmma_f16(QA[0], KB0, zero);
            Sc[t]  = wmma_f16(QA[1], KB1, s0);
        }

        // V tiles for this block: issue loads NOW so they fly behind softmax
        v16h VB[4];
#pragma unroll
        for (int nt = 0; nt < 4; nt++) {
            const _Float16* Vrow =
                Vb + (size_t)(nt * 16 + l16) * S_LEN + key0 + half * 16;
            VB[nt] = *(const v16h*)Vrow;
        }

        // prefetch next block's K and V rows into near caches (no VGPR cost)
        if (kb2 + 1 < nkb) {
            const int keyn = key0 + 32;
#pragma unroll
            for (int t = 0; t < 2; t++)
                __builtin_prefetch(Kb + (size_t)(keyn + t * 16 + l16) * HD, 0, 3);
#pragma unroll
            for (int nt = 0; nt < 4; nt++)
                __builtin_prefetch(
                    Vb + (size_t)(nt * 16 + l16) * S_LEN + keyn + half * 16, 0, 3);
        }

        // scale + causal mask + row max (C layout: row = r+8*half, col = l16)
        float rmax[8];
#pragma unroll
        for (int r = 0; r < 8; r++) {
            int q  = qt * 16 + r + half * 8;
            int k0 = key0 + l16;
            int k1 = key0 + 16 + l16;
            float v0 = (k0 <= q) ? Sc[0][r] * 0.125f : -__builtin_inff();
            float v1 = (k1 <= q) ? Sc[1][r] * 0.125f : -__builtin_inff();
            Sc[0][r] = v0; Sc[1][r] = v1;
            rmax[r] = fmaxf(v0, v1);
        }
#pragma unroll
        for (int off = 1; off < 16; off <<= 1)
#pragma unroll
            for (int r = 0; r < 8; r++)
                rmax[r] = fmaxf(rmax[r], __shfl_xor(rmax[r], off, 32));

        // online softmax update
        float alpha[8], rsum[8];
#pragma unroll
        for (int r = 0; r < 8; r++) {
            float mnew = fmaxf(mrow[r], rmax[r]);
            alpha[r] = __expf(mrow[r] - mnew);
            mrow[r] = mnew;
            float p0 = __expf(Sc[0][r] - mnew);
            float p1 = __expf(Sc[1][r] - mnew);
            Sc[0][r] = p0; Sc[1][r] = p1;
            rsum[r] = p0 + p1;
        }
#pragma unroll
        for (int off = 1; off < 16; off <<= 1)
#pragma unroll
            for (int r = 0; r < 8; r++)
                rsum[r] += __shfl_xor(rsum[r], off, 32);
#pragma unroll
        for (int r = 0; r < 8; r++) lrow[r] = lrow[r] * alpha[r] + rsum[r];
#pragma unroll
        for (int nt = 0; nt < 4; nt++)
#pragma unroll
            for (int r = 0; r < 8; r++) O[nt][r] *= alpha[r];

        // P: C layout -> LDS -> A-operand layout
#pragma unroll
        for (int t = 0; t < 2; t++)
#pragma unroll
            for (int r = 0; r < 8; r++)
                Pl[r + half * 8][t * 16 + l16] = (_Float16)Sc[t][r];
        __syncthreads();
        v16h PA;
        {
            v8h r0 = *(const v8h*)(&Pl[l16][half * 16]);
            v8h r1 = *(const v8h*)(&Pl[l16][half * 16 + 8]);
#pragma unroll
            for (int i = 0; i < 8; i++) { PA[i] = r0[i]; PA[8 + i] = r1[i]; }
        }
        __syncthreads();

        // O += P @ V  (V pre-transposed: lane = out dim, 16 contiguous keys)
#pragma unroll
        for (int nt = 0; nt < 4; nt++)
            O[nt] = wmma_f16(PA, VB[nt], O[nt]);
    }

    // normalize + store merged-head layout attn[b, s, h*64 + d] as f16
#pragma unroll
    for (int nt = 0; nt < 4; nt++)
#pragma unroll
        for (int r = 0; r < 8; r++) {
            int s = qt * 16 + r + half * 8;
            float val = O[nt][r] / lrow[r];
            attn[((size_t)(b * S_LEN + s)) * D_DIM + h * 64 + nt * 16 + l16] =
                (_Float16)val;
        }
}

// ---------------------------------------------------------------------------
// Kernel 3: output projection: out = attn @ Wo + bo  (fp32 output)
// ---------------------------------------------------------------------------
__global__ __launch_bounds__(256) void mha_proj_out(
    const _Float16* __restrict__ A, const _Float16* __restrict__ Wot,
    const float* __restrict__ bo, float* __restrict__ out)
{
    const int lane = threadIdx.x & 31;
    const int wave = threadIdx.x >> 5;
    const int half = lane >> 4;
    const int l16  = lane & 15;
    const int mBase = blockIdx.x * 256 + wave * 32;
    const int nBase = blockIdx.y * 64;

    const _Float16* Arow0 = A + (size_t)(mBase + l16) * D_DIM;
    const _Float16* Arow1 = A + (size_t)(mBase + 16 + l16) * D_DIM;
    const _Float16* Brow0 = Wot + (size_t)(nBase + 0 * 16 + l16) * D_DIM + half * 16;
    const _Float16* Brow1 = Wot + (size_t)(nBase + 1 * 16 + l16) * D_DIM + half * 16;
    const _Float16* Brow2 = Wot + (size_t)(nBase + 2 * 16 + l16) * D_DIM + half * 16;
    const _Float16* Brow3 = Wot + (size_t)(nBase + 3 * 16 + l16) * D_DIM + half * 16;

    v8f acc[2][4] = {};
    gemm_tile_32x64(Arow0, Arow1, Brow0, Brow1, Brow2, Brow3, half, acc);

#pragma unroll
    for (int nt = 0; nt < 4; nt++) {
        int gcol = nBase + nt * 16 + l16;
        float bb = bo[gcol];
#pragma unroll
        for (int mt = 0; mt < 2; mt++) {
#pragma unroll
            for (int r = 0; r < 8; r++) {
                int gm = mBase + mt * 16 + r + half * 8;
                out[(size_t)gm * D_DIM + gcol] = acc[mt][nt][r] + bb;
            }
        }
    }
}

// ---------------------------------------------------------------------------
extern "C" void kernel_launch(void* const* d_in, const int* in_sizes, int n_in,
                              void* d_out, int out_size, void* d_ws, size_t ws_size,
                              hipStream_t stream) {
    (void)in_sizes; (void)n_in; (void)out_size; (void)ws_size;
    const float* q_in = (const float*)d_in[0];
    const float* k_in = (const float*)d_in[1];
    const float* v_in = (const float*)d_in[2];
    const float* Wq   = (const float*)d_in[3];
    const float* bq   = (const float*)d_in[4];
    const float* Wk   = (const float*)d_in[5];
    const float* bk   = (const float*)d_in[6];
    const float* Wv   = (const float*)d_in[7];
    const float* bv   = (const float*)d_in[8];
    const float* Wo   = (const float*)d_in[9];
    const float* bo   = (const float*)d_in[10];
    float* out = (float*)d_out;

    const size_t WELEMS = (size_t)D_DIM * D_DIM;              // 262144
    const size_t QELEMS = (size_t)4 * H_NUM * S_LEN * HD;     // 4194304

    _Float16* Wqt  = (_Float16*)d_ws;
    _Float16* Wkt  = Wqt + WELEMS;
    _Float16* Wvt  = Wkt + WELEMS;
    _Float16* Wot  = Wvt + WELEMS;
    _Float16* Qh   = Wot + WELEMS;
    _Float16* Kh   = Qh + QELEMS;
    _Float16* Vt   = Kh + QELEMS;
    _Float16* attn = Vt + QELEMS;
    _Float16* q16  = attn + QELEMS;
    _Float16* k16  = q16 + QELEMS;
    _Float16* v16  = k16 + QELEMS;

    // Stage 0: weight transpose+convert and activation convert (independent)
    mha_prep_weights<<<4096, 256, 0, stream>>>(Wq, Wk, Wv, Wo, Wqt, Wkt, Wvt, Wot);
    mha_prep_x<<<6144, 256, 0, stream>>>(q_in, k_in, v_in, q16, k16, v16);

    // Stage 1: QKV projections (z selects Q/K/V)
    mha_proj_qkv<<<dim3(32, 8, 3), 256, 0, stream>>>(
        q16, k16, v16, Wqt, Wkt, Wvt, bq, bk, bv, Qh, Kh, Vt);

    // Stage 2: causal flash attention, one wave per (b, h, 16-query tile)
    mha_attention<<<4096, 32, 0, stream>>>(Qh, Kh, Vt, attn);

    // Stage 3: output projection
    mha_proj_out<<<dim3(32, 8), 256, 0, stream>>>(attn, Wot, bo, out);
}
